// RNN_9663676416788
// MI455X (gfx1250) — matrix operand order repaired and checked by reading, not compiled, not run on hardware
//
#include <hip/hip_runtime.h>
#include <math.h>

// Problem constants
#define B_    32
#define S_    50
#define T_    49      // S-1 timesteps
#define E_    300
#define EP_   320     // E padded to multiple of 32 (WMMA K-step)
#define H_    1024
#define G_    3072    // 3*H
#define V_    32000
#define ROWS_ (T_ * B_)   // 1568 (multiple of 32)

typedef __attribute__((ext_vector_type(16))) __bf16 v16bf;
typedef __attribute__((ext_vector_type(8)))  __bf16 v8bf;
typedef __attribute__((ext_vector_type(8)))  float  v8f;

// ---------------------------------------------------------------------------
// WMMA fragment helpers, layouts per CDNA5 ISA 7.12.2 (wave32)
// A (16x32 bf16, MxK): lane L holds row M=L&15; elements e=0..7 <-> K=8*half+e,
//                      e=8..15 <-> K=16+8*half+(e-8), half=L>>4.
// B (32x16 bf16, KxN): lane L holds col N=L&15; same K mapping as A.
// C/D (16x16 f32):     lane L, vgpr r <-> D[8*(L>>4)+r][L&15].
// All loads below are unconditional, contiguous, 16B/32B vector loads.
// ---------------------------------------------------------------------------

__device__ __forceinline__ v16bf load_A_bf(const __bf16* A, int lda, int m0, int k0) {
  int lane = threadIdx.x & 31;
  const __bf16* row = A + (size_t)(m0 + (lane & 15)) * lda;
  int h = (lane >> 4) * 8;
  v8bf lo = *(const v8bf*)(row + k0 + h);        // 16B aligned
  v8bf hi = *(const v8bf*)(row + k0 + 16 + h);   // 16B aligned
  return __builtin_shufflevector(lo, hi, 0,1,2,3,4,5,6,7,8,9,10,11,12,13,14,15);
}

__device__ __forceinline__ v16bf load_B_bf(const __bf16* W, int ldw, int n0, int k0) {
  int lane = threadIdx.x & 31;
  const __bf16* row = W + (size_t)(n0 + (lane & 15)) * ldw;
  int h = (lane >> 4) * 8;
  v8bf lo = *(const v8bf*)(row + k0 + h);
  v8bf hi = *(const v8bf*)(row + k0 + 16 + h);
  return __builtin_shufflevector(lo, hi, 0,1,2,3,4,5,6,7,8,9,10,11,12,13,14,15);
}

__device__ __forceinline__ v8f wmma_bf16(v16bf a, v16bf b, v8f c) {
  return __builtin_amdgcn_wmma_f32_16x16x32_bf16(false, a, false, b, (short)0, c, false, false);
}

__device__ __forceinline__ void store_C_bias(float* D, int ldd, int m0, int n0,
                                             v8f c, const float* bias) {
  int lane = threadIdx.x & 31;
  int n  = n0 + (lane & 15);
  int mb = m0 + (lane >> 4) * 8;
  float bv = bias[n];
#pragma unroll
  for (int r = 0; r < 8; ++r) D[(size_t)(mb + r) * ldd + n] = c[r] + bv;
}

__device__ __forceinline__ float sigmoidf_(float x) { return 1.0f / (1.0f + expf(-x)); }

// ---------------------------------------------------------------------------
// Setup kernels
// ---------------------------------------------------------------------------

__global__ void k_init_states(float* h0, float* h1, __bf16* h0b, __bf16* h1b) {
  int i = blockIdx.x * blockDim.x + threadIdx.x;
  if (i < B_ * H_) {
    h0[i] = 0.0f; h1[i] = 0.0f;
    h0b[i] = (__bf16)0.0f; h1b[i] = (__bf16)0.0f;
  }
}

// X[t*B+b][c] = bf16(wordvec[sent[b][t]][c]), zero padded cols E_..EP_-1
__global__ void k_embed(const int* __restrict__ sent, const float* __restrict__ wordvec,
                        __bf16* __restrict__ X) {
  int i = blockIdx.x * blockDim.x + threadIdx.x;
  if (i >= ROWS_ * EP_) return;
  int row = i / EP_, c = i % EP_;
  int t = row / B_, b = row % B_;
  float v = 0.0f;
  if (c < E_) v = wordvec[(size_t)sent[b * S_ + t] * E_ + c];
  X[(size_t)row * EP_ + c] = (__bf16)v;
}

__global__ void k_cvt_bf16(const float* __restrict__ src, __bf16* __restrict__ dst, int n) {
  int i = blockIdx.x * blockDim.x + threadIdx.x;
  if (i < n) dst[i] = (__bf16)src[i];
}

// W_ih0 [3072 x 300] f32 -> [3072 x 320] bf16, zero-padded
__global__ void k_cvt_pad_ih0(const float* __restrict__ src, __bf16* __restrict__ dst) {
  int i = blockIdx.x * blockDim.x + threadIdx.x;
  if (i >= G_ * EP_) return;
  int n = i / EP_, k = i % EP_;
  float v = (k < E_) ? src[(size_t)n * E_ + k] : 0.0f;
  dst[(size_t)n * EP_ + k] = (__bf16)v;
}

// ---------------------------------------------------------------------------
// GEMM kernels (all bf16 operands, f32 accumulate)
// ---------------------------------------------------------------------------

// D[M x N] = A @ W^T + bias.  MT=2 m-tiles per wave (B fragment reused).
// A bf16 [M x lda], W bf16 [N x ldw].  Mgroups = M/32.
__global__ void k_gemm_bf(const __bf16* __restrict__ A, int lda,
                          const __bf16* __restrict__ W, int ldw,
                          const float* __restrict__ bias,
                          float* __restrict__ D, int ldd,
                          int Mgroups, int Ntiles, int Ksteps) {
  int wave = blockIdx.x * (blockDim.x >> 5) + (threadIdx.x >> 5);
  if (wave >= Mgroups * Ntiles) return;
  int mg = wave / Ntiles, nt = wave % Ntiles;
  int m0 = mg * 32, n0 = nt * 16;
  v8f c0 = {}, c1 = {};
  for (int ks = 0; ks < Ksteps; ++ks) {
    int k0 = ks * 32;
    v16bf a0 = load_A_bf(A, lda, m0, k0);
    v16bf a1 = load_A_bf(A, lda, m0 + 16, k0);
    v16bf b  = load_B_bf(W, ldw, n0, k0);
    c0 = wmma_bf16(a0, b, c0);
    c1 = wmma_bf16(a1, b, c1);
  }
  store_C_bias(D, ldd, m0,      n0, c0, bias);
  store_C_bias(D, ldd, m0 + 16, n0, c1, bias);
}

// Fused pair (both depend only on previous states):
//   gh0 = h0 @ W_hh0^T + b_hh0,  gh1 = h1 @ W_hh1^T + b_hh1
// Each: M=32, N=3072 (192 n-tiles), K=1024.  384 waves total.
__global__ void k_gemm_dual(const __bf16* __restrict__ A0, const __bf16* __restrict__ W0,
                            const float* __restrict__ bias0, float* __restrict__ D0,
                            const __bf16* __restrict__ A1, const __bf16* __restrict__ W1,
                            const float* __restrict__ bias1, float* __restrict__ D1) {
  int wave = blockIdx.x * (blockDim.x >> 5) + (threadIdx.x >> 5);
  if (wave >= 384) return;
  int sel = (wave >= 192);
  int nt  = wave - (sel ? 192 : 0);
  const __bf16* A    = sel ? A1 : A0;
  const __bf16* W    = sel ? W1 : W0;
  const float*  bias = sel ? bias1 : bias0;
  float*        D    = sel ? D1 : D0;
  int n0 = nt * 16;
  v8f c0 = {}, c1 = {};
  for (int ks = 0; ks < H_ / 32; ++ks) {
    int k0 = ks * 32;
    v16bf a0 = load_A_bf(A, H_, 0,  k0);
    v16bf a1 = load_A_bf(A, H_, 16, k0);
    v16bf b  = load_B_bf(W, H_, n0, k0);
    c0 = wmma_bf16(a0, b, c0);
    c1 = wmma_bf16(a1, b, c1);
  }
  store_C_bias(D, G_, 0,  n0, c0, bias);
  store_C_bias(D, G_, 16, n0, c1, bias);
}

// ---------------------------------------------------------------------------
// Gate kernels
// ---------------------------------------------------------------------------

// Layer-0: h0 <- gate(GX0[t], GH0, h0); also refresh bf16 copy for next GEMM
__global__ void k_gates0(int t, const float* __restrict__ GX0,
                         const float* __restrict__ GH0,
                         float* __restrict__ h0, __bf16* __restrict__ h0b) {
  int i = blockIdx.x * blockDim.x + threadIdx.x;
  if (i >= B_ * H_) return;
  int b = i >> 10, j = i & (H_ - 1);
  size_t gx = (size_t)(t * B_ + b) * G_;
  size_t gh = (size_t)b * G_;
  float r = sigmoidf_(GX0[gx + j]          + GH0[gh + j]);
  float z = sigmoidf_(GX0[gx + H_ + j]     + GH0[gh + H_ + j]);
  float n = tanhf    (GX0[gx + 2 * H_ + j] + r * GH0[gh + 2 * H_ + j]);
  float hn = (1.0f - z) * n + z * h0[i];
  h0[i]  = hn;
  h0b[i] = (__bf16)hn;
}

// Layer-1: h1 <- gate(GI1, GH1, h1); stash bf16 copy + time-batched state row
__global__ void k_gates1(int t, const float* __restrict__ GI1,
                         const float* __restrict__ GH1,
                         float* __restrict__ h1, __bf16* __restrict__ h1b,
                         __bf16* __restrict__ Hs) {
  int i = blockIdx.x * blockDim.x + threadIdx.x;
  if (i >= B_ * H_) return;
  int b = i >> 10, j = i & (H_ - 1);
  size_t g = (size_t)b * G_;
  float r = sigmoidf_(GI1[g + j]          + GH1[g + j]);
  float z = sigmoidf_(GI1[g + H_ + j]     + GH1[g + H_ + j]);
  float n = tanhf    (GI1[g + 2 * H_ + j] + r * GH1[g + 2 * H_ + j]);
  float hn = (1.0f - z) * n + z * h1[i];
  h1[i]  = hn;
  h1b[i] = (__bf16)hn;
  Hs[(size_t)(t * B_ + b) * H_ + j] = (__bf16)hn;
}

// ---------------------------------------------------------------------------
// Vocab projection: logits = Hs[1568 x 1024] @ Wv^T + lin_b -> [B,T,V] order.
// Per-wave register blocking: 2 M-tiles x 8 N-tiles (32x128 out, 16 wmma/k-step).
// ---------------------------------------------------------------------------
#define MT_ 2
#define NT_ 8
__global__ void k_logits(const __bf16* __restrict__ Hs, const __bf16* __restrict__ Wv,
                         const float* __restrict__ bias, float* __restrict__ out) {
  const int MG = (ROWS_ / 16) / MT_;  // 49
  const int NG = (V_ / 16) / NT_;     // 250
  int wave = blockIdx.x * (blockDim.x >> 5) + (threadIdx.x >> 5);
  if (wave >= MG * NG) return;
  int mg = wave / NG, ng = wave % NG;
  int m0 = mg * (16 * MT_), n0 = ng * (16 * NT_);

  v8f z = {};
  v8f c[MT_][NT_];
#pragma unroll
  for (int mi = 0; mi < MT_; ++mi)
#pragma unroll
    for (int nj = 0; nj < NT_; ++nj) c[mi][nj] = z;

  for (int ks = 0; ks < H_ / 32; ++ks) {
    int k0 = ks * 32;
    v16bf a[MT_];
#pragma unroll
    for (int mi = 0; mi < MT_; ++mi) a[mi] = load_A_bf(Hs, H_, m0 + 16 * mi, k0);
#pragma unroll
    for (int nj = 0; nj < NT_; ++nj) {
      v16bf b = load_B_bf(Wv, H_, n0 + 16 * nj, k0);
#pragma unroll
      for (int mi = 0; mi < MT_; ++mi) c[mi][nj] = wmma_bf16(a[mi], b, c[mi][nj]);
    }
  }

  int lane = threadIdx.x & 31;
  int nl = lane & 15;
  int mb = (lane >> 4) * 8;
#pragma unroll
  for (int mi = 0; mi < MT_; ++mi) {
#pragma unroll
    for (int nj = 0; nj < NT_; ++nj) {
      int n = n0 + 16 * nj + nl;
      float bv = bias[n];
#pragma unroll
      for (int r = 0; r < 8; ++r) {
        int row = m0 + 16 * mi + mb + r;     // = t*B + b
        int t = row / B_, b = row % B_;
        out[1 + ((size_t)b * T_ + t) * V_ + n] = c[mi][nj][r] + bv;
      }
    }
  }
}

// ---------------------------------------------------------------------------
// Loss
// ---------------------------------------------------------------------------

// Per-row log-softmax + target gather -> tok_ll[row].  One block per (t,b).
__global__ void k_loss_rows(const float* __restrict__ out, const int* __restrict__ sent,
                            float* __restrict__ tokll) {
  int row = blockIdx.x;
  int t = row / B_, b = row % B_;
  const float* lg = out + 1 + ((size_t)b * T_ + t) * V_;
  __shared__ float red[256];
  int tid = threadIdx.x;

  float m = -INFINITY;
  for (int v = tid; v < V_; v += 256) m = fmaxf(m, lg[v]);
  red[tid] = m; __syncthreads();
  for (int s = 128; s > 0; s >>= 1) {
    if (tid < s) red[tid] = fmaxf(red[tid], red[tid + s]);
    __syncthreads();
  }
  float rowmax = red[0]; __syncthreads();

  float sum = 0.0f;
  for (int v = tid; v < V_; v += 256) sum += expf(lg[v] - rowmax);
  red[tid] = sum; __syncthreads();
  for (int s = 128; s > 0; s >>= 1) {
    if (tid < s) red[tid] += red[tid + s];
    __syncthreads();
  }
  if (tid == 0) {
    float logZ = logf(red[0]) + rowmax;
    int tgt = sent[b * S_ + t + 1];
    tokll[row] = lg[tgt] - logZ;
  }
}

// Deterministic final reduction: masked per-sample mean CE, then batch mean.
__global__ void k_loss_final(const float* __restrict__ tokll, const int* __restrict__ length,
                             float* __restrict__ out) {
  __shared__ float ps[B_];
  int b = threadIdx.x;
  if (b < B_) {
    int L = length[b];
    int ntok = (L - 1 < 1) ? 1 : (L - 1);
    float acc = 0.0f;
    for (int t = 0; t < T_; ++t)
      if (t < L - 1) acc += tokll[t * B_ + b];
    ps[b] = -acc / (float)ntok;
  }
  __syncthreads();
  if (threadIdx.x == 0) {
    float s = 0.0f;
    for (int bb = 0; bb < B_; ++bb) s += ps[bb];
    out[0] = s / (float)B_;
  }
}

// ---------------------------------------------------------------------------

extern "C" void kernel_launch(void* const* d_in, const int* in_sizes, int n_in,
                              void* d_out, int out_size, void* d_ws, size_t ws_size,
                              hipStream_t stream) {
  (void)in_sizes; (void)n_in; (void)out_size; (void)ws_size;
  const int*   sent    = (const int*)d_in[0];
  const int*   length  = (const int*)d_in[1];
  const float* wordvec = (const float*)d_in[2];
  const float* W_ih0   = (const float*)d_in[3];
  const float* W_hh0   = (const float*)d_in[4];
  const float* b_ih0   = (const float*)d_in[5];
  const float* b_hh0   = (const float*)d_in[6];
  const float* W_ih1   = (const float*)d_in[7];
  const float* W_hh1   = (const float*)d_in[8];
  const float* b_ih1   = (const float*)d_in[9];
  const float* b_hh1   = (const float*)d_in[10];
  const float* lin_W   = (const float*)d_in[11];
  const float* lin_b   = (const float*)d_in[12];
  float* out = (float*)d_out;

  // Workspace carve-out (~112 MB total)
  char* ws = (char*)d_ws;
  size_t off = 0;
  auto carve = [&](size_t bytes) -> void* {
    void* p = ws + off;
    off += (bytes + 255) & ~(size_t)255;
    return p;
  };
  __bf16* X      = (__bf16*)carve(sizeof(__bf16) * (size_t)ROWS_ * EP_);   // embeddings
  float*  GX0    = (float*)carve(sizeof(float) * (size_t)ROWS_ * G_);      // x-gates, all t
  float*  GH0    = (float*)carve(sizeof(float) * B_ * G_);
  float*  GH1    = (float*)carve(sizeof(float) * B_ * G_);
  float*  GI1    = (float*)carve(sizeof(float) * B_ * G_);
  float*  h0     = (float*)carve(sizeof(float) * B_ * H_);
  float*  h1     = (float*)carve(sizeof(float) * B_ * H_);
  __bf16* h0b    = (__bf16*)carve(sizeof(__bf16) * B_ * H_);
  __bf16* h1b    = (__bf16*)carve(sizeof(__bf16) * B_ * H_);
  __bf16* Hs     = (__bf16*)carve(sizeof(__bf16) * (size_t)ROWS_ * H_);    // h1 states
  __bf16* Wv     = (__bf16*)carve(sizeof(__bf16) * (size_t)V_ * H_);       // linear_W
  __bf16* Wb_ih0 = (__bf16*)carve(sizeof(__bf16) * (size_t)G_ * EP_);      // padded
  __bf16* Wb_hh0 = (__bf16*)carve(sizeof(__bf16) * (size_t)G_ * H_);
  __bf16* Wb_ih1 = (__bf16*)carve(sizeof(__bf16) * (size_t)G_ * H_);
  __bf16* Wb_hh1 = (__bf16*)carve(sizeof(__bf16) * (size_t)G_ * H_);
  float*  tokll  = (float*)carve(sizeof(float) * ROWS_);

  // Phase 0: init / weight conversion / time-parallel precompute
  k_init_states<<<(B_ * H_ + 255) / 256, 256, 0, stream>>>(h0, h1, h0b, h1b);
  k_embed<<<(ROWS_ * EP_ + 255) / 256, 256, 0, stream>>>(sent, wordvec, X);
  k_cvt_bf16<<<(V_ * H_ + 255) / 256, 256, 0, stream>>>(lin_W, Wv, V_ * H_);
  k_cvt_pad_ih0<<<(G_ * EP_ + 255) / 256, 256, 0, stream>>>(W_ih0, Wb_ih0);
  k_cvt_bf16<<<(G_ * H_ + 255) / 256, 256, 0, stream>>>(W_hh0, Wb_hh0, G_ * H_);
  k_cvt_bf16<<<(G_ * H_ + 255) / 256, 256, 0, stream>>>(W_ih1, Wb_ih1, G_ * H_);
  k_cvt_bf16<<<(G_ * H_ + 255) / 256, 256, 0, stream>>>(W_hh1, Wb_hh1, G_ * H_);
  {
    // GX0 = X @ W_ih0^T + b_ih0 for ALL timesteps at once (time-parallel)
    int waves = (ROWS_ / 32) * (G_ / 16);  // 49*192
    k_gemm_bf<<<(waves + 7) / 8, 256, 0, stream>>>(
        X, EP_, Wb_ih0, EP_, b_ih0, GX0, G_, ROWS_ / 32, G_ / 16, EP_ / 32);
  }

  // Phase 1: recurrence (4 kernels per timestep)
  for (int t = 0; t < T_; ++t) {
    k_gemm_dual<<<48, 256, 0, stream>>>(h0b, Wb_hh0, b_hh0, GH0,
                                        h1b, Wb_hh1, b_hh1, GH1);
    k_gates0<<<(B_ * H_ + 255) / 256, 256, 0, stream>>>(t, GX0, GH0, h0, h0b);
    k_gemm_bf<<<(192 + 7) / 8, 256, 0, stream>>>(
        h0b, H_, Wb_ih1, H_, b_ih1, GI1, G_, 1, G_ / 16, H_ / 32);
    k_gates1<<<(B_ * H_ + 255) / 256, 256, 0, stream>>>(t, GI1, GH1, h1, h1b, Hs);
  }

  // Phase 2: batched vocab projection (the 103-GFLOP GEMM) + loss
  {
    int waves = ((ROWS_ / 16) / MT_) * ((V_ / 16) / NT_);  // 49*250
    k_logits<<<(waves + 7) / 8, 256, 0, stream>>>(Hs, Wv, lin_b, out);
  }
  k_loss_rows<<<ROWS_, 256, 0, stream>>>(out, sent, tokll);
  k_loss_final<<<1, 32, 0, stream>>>(tokll, length, out);
}
